// SparseConvOutputHeadFVDB_9397388443751
// MI455X (gfx1250) — matrix operand order repaired and checked by reading, not compiled, not run on hardware
//
#include <hip/hip_runtime.h>
#include <hip/hip_bf16.h>

// MI455X / gfx1250 implementation.
// Heavy path: sparse 27-tap conv as f16 WMMA (v_wmma_f32_16x16x32_f16) with
// fp32 accumulation; per-tap B weights staged into LDS via async global->LDS
// copies (double-buffered, ASYNCcnt), fused SiLU + [128->16] WMMA head.
// Requires ws_size >= ~104 MB (1 MB tables + N*128*2 B f16 feature cache).

typedef __attribute__((ext_vector_type(16))) _Float16 v16h;
typedef __attribute__((ext_vector_type(8)))  _Float16 v8h;
typedef __attribute__((ext_vector_type(8)))  float    v8f;

union V16 { v16h v; v8h h[2]; };

#define C_IN 128
#define NGRP 8
#define EPS_GN 1e-5f
#define TAP_HALFS 16384   // 32 KB of f16 B-fragments per tap

// ---------------------------------------------------------------------------
// 1) GroupNorm statistics: per-group sum and sum-of-squares over all voxels.
// ---------------------------------------------------------------------------
__global__ __launch_bounds__(256)
void gn_stats_kernel(const float* __restrict__ feats, float* __restrict__ stats, int N)
{
    const int c    = threadIdx.x & 127;
    const int half = threadIdx.x >> 7;
    float s = 0.f, s2 = 0.f;
    for (long row = (long)blockIdx.x * 2 + half; row < N; row += (long)gridDim.x * 2) {
        float x = feats[row * C_IN + c];
        s  += x;
        s2 += x * x;
    }
    __shared__ float red[256];
    red[threadIdx.x] = s;
    __syncthreads();
    if (threadIdx.x < NGRP) {
        float a = 0.f;
        for (int i = 0; i < 16; ++i)
            a += red[threadIdx.x * 16 + i] + red[128 + threadIdx.x * 16 + i];
        atomicAdd(&stats[threadIdx.x], a);
    }
    __syncthreads();
    red[threadIdx.x] = s2;
    __syncthreads();
    if (threadIdx.x < NGRP) {
        float a = 0.f;
        for (int i = 0; i < 16; ++i)
            a += red[threadIdx.x * 16 + i] + red[128 + threadIdx.x * 16 + i];
        atomicAdd(&stats[NGRP + threadIdx.x], a);
    }
}

// ---------------------------------------------------------------------------
// 2) Fold GroupNorm into per-channel affine: y = x*scl[c] + scl[128+c]
// ---------------------------------------------------------------------------
__global__ __launch_bounds__(128)
void gn_scale_kernel(const float* __restrict__ stats,
                     const float* __restrict__ gnw, const float* __restrict__ gnb,
                     float* __restrict__ scl, int N)
{
    const int c = threadIdx.x;
    const int g = c >> 4;
    const float cnt  = (float)N * 16.0f;
    const float mean = stats[g] / cnt;
    const float var  = stats[NGRP + g] / cnt - mean * mean;
    const float s    = gnw[c] * rsqrtf(var + EPS_GN);
    scl[c]        = s;
    scl[C_IN + c] = gnb[c] - mean * s;
}

// ---------------------------------------------------------------------------
// 3) Normalize + convert to f16 feature table (L2-resident, 103 MB)
// ---------------------------------------------------------------------------
__global__ __launch_bounds__(256)
void gn_apply_kernel(const float* __restrict__ feats, const float* __restrict__ scl,
                     _Float16* __restrict__ hh, long total)
{
    long i = (long)blockIdx.x * blockDim.x + threadIdx.x;
    const long stride = (long)gridDim.x * blockDim.x;
    for (; i < total; i += stride) {
        int c = (int)(i & (C_IN - 1));
        hh[i] = (_Float16)(feats[i] * scl[c] + scl[C_IN + c]);
    }
}

// ---------------------------------------------------------------------------
// 4) Pre-swizzle conv_w [K,128,128] and lin_w [16,128] into per-lane WMMA
//    B-fragment layout (f16). B fragment for 16x16x32: lane = N column
//    (lanes 0-15 -> K rows 0-7,16-23, lanes 16-31 -> K rows 8-15,24-31), 16 halfs/lane.
//    wB flat index = ((((k*4+q)*8 + t)*32) + lane)*16 + j
// ---------------------------------------------------------------------------
__global__ __launch_bounds__(256)
void swizzle_w_kernel(const float* __restrict__ conv_w, const float* __restrict__ lin_w,
                      _Float16* __restrict__ wB, _Float16* __restrict__ wL, int K)
{
    const int id = blockIdx.x * blockDim.x + threadIdx.x;
    const int convTotal = K * 4 * 8 * 512;
    if (id < convTotal) {
        int j    = id & 15;
        int lane = (id >> 4) & 31;
        int t    = (id >> 9) & 7;
        int q    = (id >> 12) & 3;
        int k    = id >> 14;
        int kk   = (j & 7) + 8 * (lane >> 4) + 16 * (j >> 3);
        int cin  = q * 32 + kk;
        int cout = t * 16 + (lane & 15);
        wB[id] = (_Float16)conv_w[(long)k * (C_IN * C_IN) + (long)cin * C_IN + cout];
    } else if (id < convTotal + 4 * 512) {
        int id2  = id - convTotal;
        int j    = id2 & 15;
        int lane = (id2 >> 4) & 31;
        int q    = id2 >> 9;
        int kk   = (j & 7) + 8 * (lane >> 4) + 16 * (j >> 3);
        int cin  = q * 32 + kk;
        int cout = lane & 15;
        wL[(long)q * 512 + lane * 16 + j] = (_Float16)lin_w[(long)cout * C_IN + cin];
    }
}

// ---------------------------------------------------------------------------
// Async global->LDS copy of one 256 B slice per lane (16 x b128, ASYNCcnt).
// ISA adds INST_OFFSET to BOTH the global and the LDS address.
// ---------------------------------------------------------------------------
__device__ __forceinline__ void async_copy_256(unsigned ldsOff, unsigned long long gaddr)
{
    asm volatile(
        "global_load_async_to_lds_b128 %0, %1, off\n\t"
        "global_load_async_to_lds_b128 %0, %1, off offset:16\n\t"
        "global_load_async_to_lds_b128 %0, %1, off offset:32\n\t"
        "global_load_async_to_lds_b128 %0, %1, off offset:48\n\t"
        "global_load_async_to_lds_b128 %0, %1, off offset:64\n\t"
        "global_load_async_to_lds_b128 %0, %1, off offset:80\n\t"
        "global_load_async_to_lds_b128 %0, %1, off offset:96\n\t"
        "global_load_async_to_lds_b128 %0, %1, off offset:112\n\t"
        "global_load_async_to_lds_b128 %0, %1, off offset:128\n\t"
        "global_load_async_to_lds_b128 %0, %1, off offset:144\n\t"
        "global_load_async_to_lds_b128 %0, %1, off offset:160\n\t"
        "global_load_async_to_lds_b128 %0, %1, off offset:176\n\t"
        "global_load_async_to_lds_b128 %0, %1, off offset:192\n\t"
        "global_load_async_to_lds_b128 %0, %1, off offset:208\n\t"
        "global_load_async_to_lds_b128 %0, %1, off offset:224\n\t"
        "global_load_async_to_lds_b128 %0, %1, off offset:240"
        :: "v"(ldsOff), "v"(gaddr) : "memory");
}

// ---------------------------------------------------------------------------
// 5) Fused sparse conv (WMMA f16) + SiLU + linear head (WMMA f16).
//    4 waves/block; each wave: 2 M-tiles (32 rows) x 8 N-tiles (128 channels).
//    B staged per-tap into LDS via async copies, double buffered.
// ---------------------------------------------------------------------------
__global__ __launch_bounds__(128)
void conv_wmma_kernel(const _Float16* __restrict__ hh,
                      const int* __restrict__ nidx,
                      const _Float16* __restrict__ wB,
                      const _Float16* __restrict__ wL,
                      const float* __restrict__ lin_b,
                      float* __restrict__ out,
                      int K)
{
    const int lane = threadIdx.x & 31;
    const int wave = threadIdx.x >> 5;       // 0..3
    const int l15  = lane & 15;
    const int hi   = (lane >> 4) & 1;

    const long rowBase = ((long)blockIdx.x * 4 + wave) * 32;   // 2 tiles of 16 rows

    __shared__ _Float16 ldsB[2][TAP_HALFS];        // 64 KB double buffer
    __shared__ _Float16 lsilu[4][2][16 * C_IN];    // 32 KB

    // Per-thread slice of a tap block: 128 halfs = 256 B, 128 threads = 32 KB.
    const unsigned sliceHalfs = threadIdx.x * 128;
    const unsigned ldsOff0 = (unsigned)(uintptr_t)(&ldsB[0][sliceHalfs]);
    const unsigned ldsOff1 = (unsigned)(uintptr_t)(&ldsB[1][sliceHalfs]);

    // Prime tap 0 into buffer 0.
    async_copy_256(ldsOff0, (unsigned long long)(uintptr_t)(wB + sliceHalfs));

    v8f acc0[8], acc1[8];
    const v8f vzero = {0.f, 0.f, 0.f, 0.f, 0.f, 0.f, 0.f, 0.f};
#pragma unroll
    for (int t = 0; t < 8; ++t) { acc0[t] = vzero; acc1[t] = vzero; }

    for (int k = 0; k < K; ++k) {
        // Kick off tap k+1 into the other buffer, then retire tap k's copies.
        if (k + 1 < K) {
            const unsigned dst = ((k + 1) & 1) ? ldsOff1 : ldsOff0;
            async_copy_256(dst, (unsigned long long)(uintptr_t)
                                (wB + (size_t)(k + 1) * TAP_HALFS + sliceHalfs));
            asm volatile("s_wait_asynccnt 16" ::: "memory");
        } else {
            asm volatile("s_wait_asynccnt 0" ::: "memory");
        }
        __syncthreads();   // tap k fully resident in LDS for all waves

        const _Float16* tapB = &ldsB[k & 1][0];

        const int i0 = nidx[(rowBase + l15) * K + k];
        const int i1 = nidx[(rowBase + 16 + l15) * K + k];
        const _Float16* __restrict__ r0 = hh + (long)i0 * C_IN;
        const _Float16* __restrict__ r1 = hh + (long)i1 * C_IN;
#pragma unroll
        for (int q = 0; q < 4; ++q) {
            const int o = q * 32 + hi * 8;
            V16 a0, a1;
            a0.h[0] = *(const v8h*)(r0 + o);
            a0.h[1] = *(const v8h*)(r0 + o + 16);
            a1.h[0] = *(const v8h*)(r1 + o);
            a1.h[1] = *(const v8h*)(r1 + o + 16);
            const _Float16* bb = tapB + (q * 8) * 512 + lane * 16;
#pragma unroll
            for (int t = 0; t < 8; ++t) {
                v16h b = *(const v16h*)(bb + t * 512);   // 2x ds_load_b128
                acc0[t] = __builtin_amdgcn_wmma_f32_16x16x32_f16(
                    false, a0.v, false, b, (short)0, acc0[t], false, false);
                acc1[t] = __builtin_amdgcn_wmma_f32_16x16x32_f16(
                    false, a1.v, false, b, (short)0, acc1[t], false, false);
            }
        }
        __syncthreads();   // all waves done reading buf[k&1] before overwrite
    }

    // ---- SiLU, re-layout through LDS (D layout: lane=N, reg=M; head A needs
    //      lane=M, reg=K), then fused [128 -> 16] linear head via WMMA. ----
#pragma unroll
    for (int t = 0; t < 8; ++t) {
        const int ch = t * 16 + l15;
#pragma unroll
        for (int r = 0; r < 8; ++r) {
            const int M = r + 8 * hi;
            float x0 = acc0[t][r];
            float x1 = acc1[t][r];
            x0 = x0 / (1.f + __expf(-x0));         // SiLU
            x1 = x1 / (1.f + __expf(-x1));
            lsilu[wave][0][M * C_IN + ch] = (_Float16)x0;
            lsilu[wave][1][M * C_IN + ch] = (_Float16)x1;
        }
    }
    __syncthreads();

    const float bias = lin_b[l15];
#pragma unroll
    for (int m = 0; m < 2; ++m) {
        v8f hacc = vzero;
        const _Float16* tile = &lsilu[wave][m][0];
#pragma unroll
        for (int q = 0; q < 4; ++q) {
            const int o = q * 32 + hi * 8;
            V16 a;
            a.h[0] = *(const v8h*)(tile + l15 * C_IN + o);
            a.h[1] = *(const v8h*)(tile + l15 * C_IN + o + 16);
            v16h b = *(const v16h*)(wL + (long)q * 512 + lane * 16);
            hacc = __builtin_amdgcn_wmma_f32_16x16x32_f16(
                false, a.v, false, b, (short)0, hacc, false, false);
        }
        const long gr = rowBase + m * 16;
#pragma unroll
        for (int r = 0; r < 8; ++r) {
            const int M = r + 8 * hi;
            out[(gr + M) * 16 + l15] = hacc[r] + bias;
        }
    }
}

// ---------------------------------------------------------------------------
// Launcher
// ---------------------------------------------------------------------------
extern "C" void kernel_launch(void* const* d_in, const int* in_sizes, int n_in,
                              void* d_out, int out_size, void* d_ws, size_t ws_size,
                              hipStream_t stream)
{
    const float* feats  = (const float*)d_in[0];
    const int*   nidx   = (const int*)  d_in[1];
    const float* gnw    = (const float*)d_in[2];
    const float* gnb    = (const float*)d_in[3];
    const float* conv_w = (const float*)d_in[4];
    const float* lin_w  = (const float*)d_in[5];
    const float* lin_b  = (const float*)d_in[6];
    float*       out    = (float*)d_out;

    const int N = in_sizes[0] / C_IN;          // 400000
    const int K = in_sizes[1] / N;             // 27

    // Workspace layout
    char* w = (char*)d_ws;
    float*    stats = (float*)w;                                     // 64 B
    float*    scl   = (float*)(w + 64);                              // 1 KB
    _Float16* wB    = (_Float16*)(w + 4096);                         // K*32KB f16
    _Float16* wL    = (_Float16*)(w + 4096 + (size_t)K * TAP_HALFS * 2);
    _Float16* hh    = (_Float16*)(w + (1 << 20));                    // N*128 f16

    hipMemsetAsync(stats, 0, 16 * sizeof(float), stream);

    gn_stats_kernel<<<512, 256, 0, stream>>>(feats, stats, N);
    gn_scale_kernel<<<1, 128, 0, stream>>>(stats, gnw, gnb, scl, N);

    const long total = (long)N * C_IN;
    gn_apply_kernel<<<2048, 256, 0, stream>>>(feats, scl, hh, total);

    const int swTotal = K * 4 * 8 * 512 + 4 * 512;
    swizzle_w_kernel<<<(swTotal + 255) / 256, 256, 0, stream>>>(conv_w, lin_w, wB, wL, K);

    // N = 400000 = 3125 * 128 rows; 128 rows per block (4 waves x 32 rows)
    conv_wmma_kernel<<<N / 128, 128, 0, stream>>>(hh, nidx, wB, wL, lin_b, out, K);
}
